// DGCNN_48696339202751
// MI455X (gfx1250) — matrix operand order repaired and checked by reading, not compiled, not run on hardware
//
#include <hip/hip_runtime.h>
#include <stdint.h>

// ---------------- CDNA5 / gfx1250 WMMA plumbing ----------------
typedef __attribute__((ext_vector_type(16))) __bf16 v16bf;
typedef __attribute__((ext_vector_type(8)))  float  v8f;
typedef __attribute__((ext_vector_type(8)))  unsigned int v8u;

#define K_NN    20
#define BATCH   8
#define NPTS    2048
#define BN_TOT  (BATCH * NPTS)
#define FSTRIDE 1536
#define EPSBN   1e-5f
#define SLOPE   0.2f

__device__ __forceinline__ v8f wmma_bf16(v8u a, v8u b, v8f c) {
  return __builtin_amdgcn_wmma_f32_16x16x32_bf16(
      false, __builtin_bit_cast(v16bf, a),
      false, __builtin_bit_cast(v16bf, b),
      (short)0, c, false, false);
}

// bf16 pack/unpack (manual RNE; storage is uint16_t)
__device__ __forceinline__ uint16_t f2bf(float f) {
  uint32_t u = __float_as_uint(f);
  return (uint16_t)((u + 0x7FFFu + ((u >> 16) & 1u)) >> 16);
}
__device__ __forceinline__ float bf2f(uint16_t u) {
  return __uint_as_float(((uint32_t)u) << 16);
}
__device__ __forceinline__ float bflo(uint32_t u) { return __uint_as_float((u & 0xFFFFu) << 16); }
__device__ __forceinline__ float bfhi(uint32_t u) { return __uint_as_float(u & 0xFFFF0000u); }

// order-preserving float<->uint key for atomicMax/Min on floats
__device__ __forceinline__ unsigned f2key(float f) {
  unsigned u = __float_as_uint(f);
  return (u & 0x80000000u) ? ~u : (u | 0x80000000u);
}
__device__ __forceinline__ float key2f(unsigned k) {
  unsigned u = (k & 0x80000000u) ? (k & 0x7FFFFFFFu) : ~k;
  return __uint_as_float(u);
}
__device__ __forceinline__ float lrelu(float x) { return x >= 0.f ? x : SLOPE * x; }

// A-fragment (16x32 bf16) K-pair index for VGPR v, half 'khalf' (contiguous k = 16*khalf + 2v, matches B)
__device__ __forceinline__ int a_kpair(int v, int khalf) {
  return 2 * v + 16 * khalf;
}

// ================= kNN: WMMA Gram tiles + register top-k =================
// block = 1 wave (32 threads), owns 32 consecutive rows of one batch.
__global__ __launch_bounds__(32) void knn_kernel(
    const uint16_t* __restrict__ feat, int stride, int choff, int cst,
    const float* __restrict__ sq, int* __restrict__ idx)
{
  __shared__ uint32_t sfeat[32 * 128];   // 32 rows x up to 256ch (u32 pairs)
  __shared__ float sdist[32][16];
  const int b = blockIdx.y;
  const int r0 = blockIdx.x * 32;
  const int lane = threadIdx.x;
  const int cpair = cst >> 1;
  const uint16_t* fb = feat + (size_t)(b * NPTS) * stride + choff;

  for (int i = lane; i < 32 * cpair; i += 32) {
    int row = i / cpair, c = i - row * cpair;
    sfeat[row * cpair + c] = *(const uint32_t*)(fb + (size_t)(r0 + row) * stride + 2 * c);
  }
  __syncthreads();

  float dk[K_NN]; int ik[K_NN];
#pragma unroll
  for (int t = 0; t < K_NN; ++t) { dk[t] = -3.0e38f; ik[t] = 0; }

  const int mlane = lane & 15, khalf = lane >> 4;
  const int chunks = cst >> 5;

  for (int ct = 0; ct < NPTS / 16; ++ct) {
    const int col = ct * 16 + mlane;
    const uint16_t* fc = fb + (size_t)col * stride;
    if (ct + 1 < NPTS / 16)                 // cover next tile's gather latency
      __builtin_prefetch(fc + (size_t)16 * stride, 0, 1);
    const float sqc = sq[b * NPTS + col];
    v8f acc0 = {}; v8f acc1 = {};
    for (int ch = 0; ch < chunks; ++ch) {
      const int kb = ch * 32;
      v8u bv, a0, a1;
#pragma unroll
      for (int v = 0; v < 8; ++v)
        bv[v] = *(const uint32_t*)(fc + kb + 2 * v + 16 * khalf);
#pragma unroll
      for (int v = 0; v < 8; ++v) {
        const int kp = (kb + a_kpair(v, khalf)) >> 1;
        a0[v] = sfeat[mlane * cpair + kp];
        a1[v] = sfeat[(16 + mlane) * cpair + kp];
      }
      acc0 = wmma_bf16(a0, bv, acc0);
      acc1 = wmma_bf16(a1, bv, acc1);
    }
#pragma unroll
    for (int v = 0; v < 8; ++v) {
      const int rl = v + 8 * khalf;
      sdist[rl][mlane]      = 2.0f * acc0[v] - sqc;
      sdist[16 + rl][mlane] = 2.0f * acc1[v] - sqc;
    }
    __syncthreads();
#pragma unroll 4
    for (int i = 0; i < 16; ++i) {
      float d = sdist[lane][i];
      if (d > dk[K_NN - 1]) {
        float cd = d; int ci = ct * 16 + i;
#pragma unroll
        for (int t = 0; t < K_NN; ++t) {
          bool sw = cd > dk[t];
          float td = dk[t]; int ti = ik[t];
          dk[t] = sw ? cd : td; ik[t] = sw ? ci : ti;
          cd = sw ? td : cd;    ci = sw ? ti : ci;
        }
      }
    }
    __syncthreads();
  }
  int* op = idx + (size_t)(b * NPTS + r0 + lane) * K_NN;
#pragma unroll
  for (int t = 0; t < K_NN; ++t) op[t] = ik[t];
}

// ============ EdgeConv GEMM, edge fragments register-cached across O-tiles ===
// CHUNKS = Cedge/32 (even); first CHUNKS/2 chunks are the (feat[p]-feat[n])
// half, rest are the center copy half -> no per-lane divergence.
template <int CHUNKS, int OTILES>
__global__ __launch_bounds__(32) void edge_conv_t(
    const uint16_t* __restrict__ feat, int stride, int choff,
    const uint16_t* __restrict__ wp,
    const int* __restrict__ idx,
    unsigned* __restrict__ maxk, unsigned* __restrict__ mink,
    float* __restrict__ sums, float* __restrict__ sumsq)
{
  constexpr int CST = CHUNKS * 16;
  constexpr int CEDGE = CHUNKS * 32;
  constexpr int O = OTILES * 16;
  const int b = blockIdx.z, ct = blockIdx.x;
  const int lane = threadIdx.x, mlane = lane & 15, khalf = lane >> 4;
  const int col = ct * 16 + mlane;
  const int n = col / K_NN, j = col - n * K_NN;
  const int p = idx[(size_t)(b * NPTS + n) * K_NN + j];
  const uint16_t* fb = feat + (size_t)(b * NPTS) * stride + choff;
  const uint16_t* fn = fb + (size_t)n * stride;
  const uint16_t* fp = fb + (size_t)p * stride;

  // build edge-feature B fragments once, keep in VGPRs
  v8u bfrag[CHUNKS];
#pragma unroll
  for (int ch = 0; ch < CHUNKS; ++ch) {
    const int kb = ch * 32;
#pragma unroll
    for (int v = 0; v < 8; ++v) {
      const int c = kb + 2 * v + 16 * khalf;
      if (ch < CHUNKS / 2) {                // diff half
        uint32_t up = *(const uint32_t*)(fp + c);
        uint32_t un = *(const uint32_t*)(fn + c);
        bfrag[ch][v] = (uint32_t)f2bf(bflo(up) - bflo(un)) |
                       ((uint32_t)f2bf(bfhi(up) - bfhi(un)) << 16);
      } else {                              // center half: straight copy
        bfrag[ch][v] = *(const uint32_t*)(fn + (c - CST));
      }
    }
  }

  for (int ot = 0; ot < OTILES; ++ot) {
    const uint16_t* wr = wp + (size_t)(ot * 16 + mlane) * CEDGE;
    v8f acc = {};
#pragma unroll
    for (int ch = 0; ch < CHUNKS; ++ch) {
      v8u a;
#pragma unroll
      for (int v = 0; v < 8; ++v)
        a[v] = *(const uint32_t*)(wr + ch * 32 + a_kpair(v, khalf));
      acc = wmma_bf16(a, bfrag[ch], acc);
    }
    // per-(b,o,n) running max/min over k (ordered-uint atomics)
#pragma unroll
    for (int v = 0; v < 8; ++v) {
      const int o = ot * 16 + v + 8 * khalf;
      const size_t key = (size_t)(b * O + o) * NPTS + n;
      unsigned kk = f2key(acc[v]);
      atomicMax(&maxk[key], kk);
      atomicMin(&mink[key], kk);
    }
    // per-channel sum / sumsq (half-wave tree + atomics)
#pragma unroll
    for (int v = 0; v < 8; ++v) {
      float s = acc[v], q = acc[v] * acc[v];
      for (int m = 8; m >= 1; m >>= 1) { s += __shfl_xor(s, m, 32); q += __shfl_xor(q, m, 32); }
      if (mlane == 0) {
        const int o = ot * 16 + v + 8 * khalf;
        atomicAdd(&sums[o], s); atomicAdd(&sumsq[o], q);
      }
    }
  }
}

// ============ 1x1 conv GEMM over BN columns ============
// mode 0: per-(b,o) max/min over n; mode 1: store raw bf16 [o][bn]
__global__ __launch_bounds__(32) void lin_conv_kernel(
    const uint16_t* __restrict__ feat, int stride, int choff, int Cin,
    const uint16_t* __restrict__ wp, int O, int mode,
    unsigned* __restrict__ maxk, unsigned* __restrict__ mink,
    uint16_t* __restrict__ rawout,
    float* __restrict__ sums, float* __restrict__ sumsq)
{
  const int ot = blockIdx.y, ct = blockIdx.x;
  const int lane = threadIdx.x, mlane = lane & 15, khalf = lane >> 4;
  const int bn = ct * 16 + mlane;
  const uint16_t* fc = feat + (size_t)bn * stride + choff;
  const uint16_t* wr = wp + (size_t)(ot * 16 + mlane) * Cin;

  v8f acc = {};
  for (int ch = 0; ch < (Cin >> 5); ++ch) {
    const int kb = ch * 32;
    v8u a, bv;
#pragma unroll
    for (int v = 0; v < 8; ++v)
      a[v] = *(const uint32_t*)(wr + kb + a_kpair(v, khalf));
#pragma unroll
    for (int v = 0; v < 8; ++v)
      bv[v] = *(const uint32_t*)(fc + kb + 2 * v + 16 * khalf);
    acc = wmma_bf16(a, bv, acc);
  }

  if (mode == 0) {
    const int bb = bn / NPTS;
#pragma unroll
    for (int v = 0; v < 8; ++v) {
      float mx = acc[v], mn = acc[v];
      for (int m = 8; m >= 1; m >>= 1) {
        mx = fmaxf(mx, __shfl_xor(mx, m, 32));
        mn = fminf(mn, __shfl_xor(mn, m, 32));
      }
      if (mlane == 0) {
        const int o = ot * 16 + v + 8 * khalf;
        atomicMax(&maxk[bb * O + o], f2key(mx));
        atomicMin(&mink[bb * O + o], f2key(mn));
      }
    }
  } else {
#pragma unroll
    for (int v = 0; v < 8; ++v) {
      const int o = ot * 16 + v + 8 * khalf;
      rawout[(size_t)o * BN_TOT + bn] = f2bf(acc[v]);
    }
  }
#pragma unroll
  for (int v = 0; v < 8; ++v) {
    float s = acc[v], q = acc[v] * acc[v];
    for (int m = 8; m >= 1; m >>= 1) { s += __shfl_xor(s, m, 32); q += __shfl_xor(q, m, 32); }
    if (mlane == 0) {
      const int o = ot * 16 + v + 8 * khalf;
      atomicAdd(&sums[o], s); atomicAdd(&sumsq[o], q);
    }
  }
}

// ================= small elementwise kernels =================
__global__ void finalize_stats_kernel(const float* sums, const float* sumsq,
                                      const float* g, const float* bb, int O,
                                      float invcnt, float* scale, float* shift)
{
  int o = blockIdx.x * blockDim.x + threadIdx.x;
  if (o >= O) return;
  float m = sums[o] * invcnt;
  float v = sumsq[o] * invcnt - m * m;
  v = v < 0.f ? 0.f : v;
  float sc = g[o] * rsqrtf(v + EPSBN);
  scale[o] = sc;
  shift[o] = bb[o] - m * sc;
}

// BN+lrelu on per-(b,o,n) extremum -> feature buffer channel slot
__global__ void apply_edge_kernel(const unsigned* maxk, const unsigned* mink,
                                  const float* scale, const float* shift,
                                  int O, int choff, uint16_t* F)
{
  int t = blockIdx.x * blockDim.x + threadIdx.x;
  if (t >= BATCH * O * NPTS) return;
  int n = t % NPTS; int bo = t / NPTS; int o = bo % O; int b = bo / O;
  float sc = scale[o];
  float sel = key2f(sc >= 0.f ? maxk[t] : mink[t]);
  float h = lrelu(sc * sel + shift[o]);
  F[(size_t)(b * NPTS + n) * FSTRIDE + choff + o] = f2bf(h);
}

__global__ void sqnorm_kernel(const uint16_t* F, int stride, int choff, int C, float* sq)
{
  int t = blockIdx.x * blockDim.x + threadIdx.x;
  if (t >= BN_TOT) return;
  const uint16_t* f = F + (size_t)t * stride + choff;
  float s = 0.f;
  for (int c = 0; c < C; ++c) { float x = bf2f(f[c]); s += x * x; }
  sq[t] = s;
}

__global__ void apply_hg_kernel(const unsigned* maxk, const unsigned* mink,
                                const float* scale, const float* shift, uint16_t* F)
{
  int t = blockIdx.x * blockDim.x + threadIdx.x;
  if (t >= BATCH * 1024) return;
  int o = t & 1023, b = t >> 10;
  float sc = scale[o];
  float sel = key2f(sc >= 0.f ? maxk[t] : mink[t]);
  uint16_t hb = f2bf(lrelu(sc * sel + shift[o]));
  uint16_t* dst = F + (size_t)(b * NPTS) * FSTRIDE + 512 + o;
  for (int n = 0; n < NPTS; ++n) dst[(size_t)n * FSTRIDE] = hb;
}

__global__ void apply_conv6_kernel(const uint16_t* raw, const float* scale,
                                   const float* shift, uint16_t* act)
{
  int t = blockIdx.x * blockDim.x + threadIdx.x;
  if (t >= BN_TOT * 512) return;
  int o = t & 511; int bn = t >> 9;
  float h = lrelu(scale[o] * bf2f(raw[(size_t)o * BN_TOT + bn]) + shift[o]);
  act[t] = f2bf(h);
}

__global__ void finalize_out_kernel(const unsigned* maxk, const unsigned* mink,
                                    const float* scale, const float* shift, float* out)
{
  int t = blockIdx.x * blockDim.x + threadIdx.x;
  if (t >= BATCH * 256) return;
  int o = t & 255;
  float sc = scale[o];
  float sel = key2f(sc >= 0.f ? maxk[t] : mink[t]);
  out[t] = lrelu(sc * sel + shift[o]);
}

// ================= weight / input prep =================
__global__ void prep_input_kernel(const float* x, uint16_t* fb0, float* sq)
{
  int t = blockIdx.x * blockDim.x + threadIdx.x;
  if (t >= BN_TOT) return;
  float s = 0.f;
  for (int c = 0; c < 32; ++c) {
    float v = (c < 3) ? x[(size_t)t * 3 + c] : 0.f;
    uint16_t h = f2bf(v);
    fb0[(size_t)t * 32 + c] = h;
    float vr = bf2f(h); s += vr * vr;
  }
  sq[t] = s;
}

// pad/reorder edge weights (O x 2*Cin f32) -> (O x 2*cst bf16)
__global__ void prep_w_edge_kernel(const float* w, int O, int Cin, int cst, uint16_t* wp)
{
  int t = blockIdx.x * blockDim.x + threadIdx.x;
  int cedge = 2 * cst;
  if (t >= O * cedge) return;
  int o = t / cedge, c = t - o * cedge;
  float val = 0.f;
  if (c < cst) { if (c < Cin) val = w[(size_t)o * 2 * Cin + c]; }
  else { int c2 = c - cst; if (c2 < Cin) val = w[(size_t)o * 2 * Cin + Cin + c2]; }
  wp[t] = f2bf(val);
}

__global__ void prep_w_copy_kernel(const float* w, int total, uint16_t* wp)
{
  int t = blockIdx.x * blockDim.x + threadIdx.x;
  if (t >= total) return;
  wp[t] = f2bf(w[t]);
}

// w6 expects [hg(1024) | x(512)]; our buffer is [x(512) | hg(1024)]
__global__ void prep_w6_kernel(const float* w6, uint16_t* wp)
{
  int t = blockIdx.x * blockDim.x + threadIdx.x;
  if (t >= 512 * 1536) return;
  int o = t / 1536, c = t - o * 1536;
  float v = (c < 512) ? w6[(size_t)o * 1536 + 1024 + c]
                      : w6[(size_t)o * 1536 + (c - 512)];
  wp[t] = f2bf(v);
}

// ================= host driver =================
extern "C" void kernel_launch(void* const* d_in, const int* in_sizes, int n_in,
                              void* d_out, int out_size, void* d_ws, size_t ws_size,
                              hipStream_t stream)
{
  const float* x  = (const float*)d_in[0];
  const float* w[8]; const float* g[8]; const float* bb[8];
  for (int i = 1; i <= 7; ++i) { w[i] = (const float*)d_in[i];
                                 g[i] = (const float*)d_in[7 + i];
                                 bb[i] = (const float*)d_in[14 + i]; }

  char* wsp = (char*)d_ws;
  auto alloc = [&](size_t bytes) { void* p = wsp; wsp += (bytes + 255) & ~(size_t)255; return p; };

  uint16_t* F     = (uint16_t*)alloc((size_t)BN_TOT * FSTRIDE * 2);
  uint16_t* fb0   = (uint16_t*)alloc((size_t)BN_TOT * 32 * 2);
  float*    sq    = (float*)   alloc((size_t)BN_TOT * 4);
  int*      idx   = (int*)     alloc((size_t)BN_TOT * K_NN * 4);
  unsigned* maxk  = (unsigned*)alloc((size_t)BATCH * 256 * NPTS * 4);
  unsigned* mink  = (unsigned*)alloc((size_t)BATCH * 256 * NPTS * 4);
  unsigned* mk5   = (unsigned*)alloc((size_t)BATCH * 1024 * 4);
  unsigned* mn5   = (unsigned*)alloc((size_t)BATCH * 1024 * 4);
  unsigned* mk7   = (unsigned*)alloc((size_t)BATCH * 256 * 4);
  unsigned* mn7   = (unsigned*)alloc((size_t)BATCH * 256 * 4);
  float*    sums  = (float*)   alloc(1024 * 4);
  float*    sumsq = (float*)   alloc(1024 * 4);
  float*    scale = (float*)   alloc(1024 * 4);
  float*    shift = (float*)   alloc(1024 * 4);
  uint16_t* wp1   = (uint16_t*)alloc((size_t)64 * 64 * 2);
  uint16_t* wp2   = (uint16_t*)alloc((size_t)64 * 128 * 2);
  uint16_t* wp3   = (uint16_t*)alloc((size_t)128 * 256 * 2);
  uint16_t* wp4   = (uint16_t*)alloc((size_t)256 * 512 * 2);
  uint16_t* wp5   = (uint16_t*)alloc((size_t)1024 * 512 * 2);
  uint16_t* wp6   = (uint16_t*)alloc((size_t)512 * 1536 * 2);
  uint16_t* wp7   = (uint16_t*)alloc((size_t)256 * 512 * 2);
  uint16_t* h6raw = (uint16_t*)alloc((size_t)512 * BN_TOT * 2);
  uint16_t* h6act = (uint16_t*)alloc((size_t)BN_TOT * 512 * 2);

  const int TB = 256;
  auto blks = [](int n, int t) { return (n + t - 1) / t; };

  // ---- prep ----
  prep_input_kernel<<<blks(BN_TOT, TB), TB, 0, stream>>>(x, fb0, sq);
  prep_w_edge_kernel<<<blks(64 * 64, TB), TB, 0, stream>>>(w[1], 64, 3, 32, wp1);
  prep_w_edge_kernel<<<blks(64 * 128, TB), TB, 0, stream>>>(w[2], 64, 64, 64, wp2);
  prep_w_edge_kernel<<<blks(128 * 256, TB), TB, 0, stream>>>(w[3], 128, 64, 64, wp3);
  prep_w_edge_kernel<<<blks(256 * 512, TB), TB, 0, stream>>>(w[4], 256, 128, 128, wp4);
  prep_w_copy_kernel<<<blks(1024 * 512, TB), TB, 0, stream>>>(w[5], 1024 * 512, wp5);
  prep_w6_kernel<<<blks(512 * 1536, TB), TB, 0, stream>>>(w[6], wp6);
  prep_w_copy_kernel<<<blks(256 * 512, TB), TB, 0, stream>>>(w[7], 256 * 512, wp7);

  // ---- edge conv layers ----
  struct Layer { const uint16_t* fin; int stride, choff, cst, O, outoff;
                 const uint16_t* wp; const float *gg, *bbp; };
  Layer L[4] = {
    { fb0, 32,   0,   32,  64,  0,   wp1, g[1], bb[1] },
    { F, FSTRIDE, 0,  64,  64,  64,  wp2, g[2], bb[2] },
    { F, FSTRIDE, 64, 64,  128, 128, wp3, g[3], bb[3] },
    { F, FSTRIDE, 128,128, 256, 256, wp4, g[4], bb[4] },
  };
  const float invcnt2d = 1.f / (float)(BATCH * NPTS * K_NN);
  const dim3 egrid(NPTS * K_NN / 16, 1, BATCH);
  for (int li = 0; li < 4; ++li) {
    Layer& l = L[li];
    knn_kernel<<<dim3(NPTS / 32, BATCH), 32, 0, stream>>>(l.fin, l.stride, l.choff, l.cst, sq, idx);
    hipMemsetAsync(sums, 0, 1024 * 4, stream);
    hipMemsetAsync(sumsq, 0, 1024 * 4, stream);
    hipMemsetAsync(maxk, 0x00, (size_t)BATCH * l.O * NPTS * 4, stream);
    hipMemsetAsync(mink, 0xFF, (size_t)BATCH * l.O * NPTS * 4, stream);
    switch (li) {
      case 0: edge_conv_t<2, 4 ><<<egrid, 32, 0, stream>>>(l.fin, l.stride, l.choff, l.wp, idx, maxk, mink, sums, sumsq); break;
      case 1: edge_conv_t<4, 4 ><<<egrid, 32, 0, stream>>>(l.fin, l.stride, l.choff, l.wp, idx, maxk, mink, sums, sumsq); break;
      case 2: edge_conv_t<4, 8 ><<<egrid, 32, 0, stream>>>(l.fin, l.stride, l.choff, l.wp, idx, maxk, mink, sums, sumsq); break;
      default:edge_conv_t<8, 16><<<egrid, 32, 0, stream>>>(l.fin, l.stride, l.choff, l.wp, idx, maxk, mink, sums, sumsq); break;
    }
    finalize_stats_kernel<<<blks(l.O, TB), TB, 0, stream>>>(sums, sumsq, l.gg, l.bbp, l.O, invcnt2d, scale, shift);
    apply_edge_kernel<<<blks(BATCH * l.O * NPTS, TB), TB, 0, stream>>>(maxk, mink, scale, shift, l.O, l.outoff, F);
    if (li < 3)
      sqnorm_kernel<<<blks(BN_TOT, TB), TB, 0, stream>>>(F, FSTRIDE, l.outoff, l.O, sq);
  }

  const float invcnt1d = 1.f / (float)BN_TOT;
  // ---- conv5 (only its N-max survives as hg) ----
  hipMemsetAsync(sums, 0, 1024 * 4, stream);
  hipMemsetAsync(sumsq, 0, 1024 * 4, stream);
  hipMemsetAsync(mk5, 0x00, (size_t)BATCH * 1024 * 4, stream);
  hipMemsetAsync(mn5, 0xFF, (size_t)BATCH * 1024 * 4, stream);
  lin_conv_kernel<<<dim3(BN_TOT / 16, 1024 / 16), 32, 0, stream>>>(
      F, FSTRIDE, 0, 512, wp5, 1024, 0, mk5, mn5, nullptr, sums, sumsq);
  finalize_stats_kernel<<<blks(1024, TB), TB, 0, stream>>>(sums, sumsq, g[5], bb[5], 1024, invcnt1d, scale, shift);
  apply_hg_kernel<<<blks(BATCH * 1024, TB), TB, 0, stream>>>(mk5, mn5, scale, shift, F);

  // ---- conv6 (full map needed) ----
  hipMemsetAsync(sums, 0, 1024 * 4, stream);
  hipMemsetAsync(sumsq, 0, 1024 * 4, stream);
  lin_conv_kernel<<<dim3(BN_TOT / 16, 512 / 16), 32, 0, stream>>>(
      F, FSTRIDE, 0, 1536, wp6, 512, 1, nullptr, nullptr, h6raw, sums, sumsq);
  finalize_stats_kernel<<<blks(512, TB), TB, 0, stream>>>(sums, sumsq, g[6], bb[6], 512, invcnt1d, scale, shift);
  apply_conv6_kernel<<<blks(BN_TOT * 512, TB), TB, 0, stream>>>(h6raw, scale, shift, h6act);

  // ---- conv7 + global max -> output ----
  hipMemsetAsync(sums, 0, 1024 * 4, stream);
  hipMemsetAsync(sumsq, 0, 1024 * 4, stream);
  hipMemsetAsync(mk7, 0x00, (size_t)BATCH * 256 * 4, stream);
  hipMemsetAsync(mn7, 0xFF, (size_t)BATCH * 256 * 4, stream);
  lin_conv_kernel<<<dim3(BN_TOT / 16, 256 / 16), 32, 0, stream>>>(
      h6act, 512, 0, 512, wp7, 256, 0, mk7, mn7, nullptr, sums, sumsq);
  finalize_stats_kernel<<<blks(256, TB), TB, 0, stream>>>(sums, sumsq, g[7], bb[7], 256, invcnt1d, scale, shift);
  finalize_out_kernel<<<blks(BATCH * 256, TB), TB, 0, stream>>>(mk7, mn7, scale, shift, (float*)d_out);
}
